// SelfAttn_55671366091221
// MI455X (gfx1250) — matrix operand order repaired and checked by reading, not compiled, not run on hardware
//
#include <hip/hip_runtime.h>
#include <hip/hip_bf16.h>
#include <stdint.h>

#define BATCH 4
#define CH    256
#define CQ    32
#define NPIX  4096

typedef __bf16 bf16_t;
typedef __attribute__((ext_vector_type(16))) __bf16 v16bf;
typedef __attribute__((ext_vector_type(8)))  __bf16 v8bf;
typedef __attribute__((ext_vector_type(8)))  float  v8f;
typedef __attribute__((ext_vector_type(4)))  float  v4f;
typedef __attribute__((ext_vector_type(4)))  unsigned int u32x4;
typedef __attribute__((ext_vector_type(8)))  int i32x8;
typedef __attribute__((ext_vector_type(4)))  int i32x4;

#if __has_builtin(__builtin_amdgcn_tensor_load_to_lds)
#define HAVE_TDM 1
#else
#define HAVE_TDM 0
#endif

// Native f32 -> bf16 (RNE); gfx1250 has hardware bf16 converts.
static __device__ __forceinline__ bf16_t f2bf(float f) { return (bf16_t)f; }

static __device__ __forceinline__ v8f zero8() {
  v8f z;
#pragma unroll
  for (int i = 0; i < 8; ++i) z[i] = 0.0f;
  return z;
}

static __device__ __forceinline__ v16bf cat16(v8bf a, v8bf b) {
  v16bf r;
#pragma unroll
  for (int i = 0; i < 8; ++i) { r[i] = a[i]; r[i + 8] = b[i]; }
  return r;
}

// A fragment (16x32, bf16): source row-major, row stride ld elements.
// Lane l (lr=l&15, hi=l>>4) holds row lr; K = hi*16 + 0..15
// (two contiguous 16B loads).
static __device__ __forceinline__ v16bf load_a_rm(const bf16_t* base, int ld, int lane) {
  int lr = lane & 15, hi = lane >> 4;
  const bf16_t* p = base + (size_t)lr * ld + hi * 16;
  v8bf lo  = *reinterpret_cast<const v8bf*>(p);
  v8bf hi8 = *reinterpret_cast<const v8bf*>(p + 8);
  return cat16(lo, hi8);
}

// B fragment (32x16, bf16): source layout S[col][k], k contiguous, stride ld.
// Lane l holds column lr, K = hi*16 + 0..15 (two contiguous 16B loads).
static __device__ __forceinline__ v16bf load_b_cm(const bf16_t* base, int ld, int lane) {
  int lr = lane & 15, hi = lane >> 4;
  const bf16_t* p = base + (size_t)lr * ld + hi * 16;
  v8bf lo  = *reinterpret_cast<const v8bf*>(p);
  v8bf hi8 = *reinterpret_cast<const v8bf*>(p + 8);
  return cat16(lo, hi8);
}

// B fragment from fp32 LDS tile (layout S[col][k], stride ld floats), with
// native convert to bf16 at fragment-build time (co-executes with WMMA).
static __device__ __forceinline__ v16bf load_b_cm_f32(const float* base, int ld, int lane) {
  int lr = lane & 15, hi = lane >> 4;
  const float* p = base + (size_t)lr * ld + hi * 16;
  v16bf r;
#pragma unroll
  for (int i = 0; i < 16; ++i) r[i] = (bf16_t)p[i];
  return r;
}

#define WMMA_BF16(A, B, C) \
  __builtin_amdgcn_wmma_f32_16x16x32_bf16(false, (A), false, (B), (short)0, (C), false, false)

#if HAVE_TDM
// TDM: DMA a 32x32 fp32 tile (row stride NPIX floats) from global into LDS,
// padding each 32-DWORD row with 4 DWORDs -> LDS row stride = 36 floats
// (bank-conflict-free for the fragment read pattern). D# per ISA 08 §8.
// This toolchain's builtin takes the 6-arg form:
//   (u32x4 g0, i32x8 g1, i32x4 g2, i32x4 g3, i32x8 extra, i32 cpol)
static __device__ __forceinline__ void tdm_load_tile(const float* gsrc, unsigned lds_off) {
  unsigned long long ga = (unsigned long long)(size_t)gsrc;
  u32x4 g0;
  g0[0] = 1u;                                             // count=1, user descriptor
  g0[1] = lds_off;                                        // lds_addr (bytes)
  g0[2] = (unsigned)(ga & 0xFFFFFFFFu);                   // global_addr[31:0]
  g0[3] = (unsigned)((ga >> 32) & 0x01FFFFFFu) | (2u << 30); // addr[56:32] | type=2
  i32x8 g1;
  g1[0] = (int)((2u << 16)      // data_size = 4 bytes
              | (1u << 20)      // pad_enable
              | (4u << 22)      // pad_interval: 2^(4+1)=32 DWORDs
              | (3u << 25));    // pad_amount: 3+1 = 4 DWORDs
  g1[1] = (int)((unsigned)NPIX << 16);  // tensor_dim0 low16 (=4096)
  g1[2] = (int)((unsigned)NPIX << 16);  // dim0 hi16(0) | tensor_dim1 low16
  g1[3] = (int)(32u << 16);             // dim1 hi16(0) | tile_dim0 = 32
  g1[4] = 32;                           // tile_dim1 = 32 | tile_dim2 = 0
  g1[5] = NPIX;                         // tensor_dim0_stride low32 = 4096
  g1[6] = 0;                            // stride0 hi16 | stride1 low16
  g1[7] = 0;
  i32x4 z4; z4[0] = 0; z4[1] = 0; z4[2] = 0; z4[3] = 0;
  i32x8 z8;
#pragma unroll
  for (int i = 0; i < 8; ++i) z8[i] = 0;
  __builtin_amdgcn_tensor_load_to_lds(g0, g1, z4, z4, z8, 0);
}
#endif

// ---------------------------------------------------------------- conversions
__global__ void cvt_f32_bf16(const float* __restrict__ src, bf16_t* __restrict__ dst, int n) {
  int i = blockIdx.x * blockDim.x + threadIdx.x;
  if (i < n) dst[i] = f2bf(src[i]);
}

// x (B,C,N) f32 -> xT (B,N,C) bf16 via LDS-tiled transpose
__global__ void __launch_bounds__(256) transpose_cvt(const float* __restrict__ x,
                                                     bf16_t* __restrict__ xT) {
  __shared__ float tile[32][33];
  int b  = blockIdx.z;
  int c0 = blockIdx.y * 32;
  int n0 = blockIdx.x * 32;
  int tx = threadIdx.x, ty = threadIdx.y;   // 32 x 8
  const float* xb = x + (size_t)b * CH * NPIX;
  for (int i = ty; i < 32; i += 8)
    tile[i][tx] = xb[(size_t)(c0 + i) * NPIX + n0 + tx];
  __syncthreads();
  bf16_t* xTb = xT + (size_t)b * NPIX * CH;
  for (int i = ty; i < 32; i += 8)
    xTb[(size_t)(n0 + i) * CH + c0 + tx] = f2bf(tile[tx][i]);
}

// ---------------------------------------------------------------- q/k proj
__global__ void __launch_bounds__(256) proj_qk(const bf16_t* __restrict__ xT,
                                               const bf16_t* __restrict__ wq,
                                               const bf16_t* __restrict__ wk,
                                               const float* __restrict__ bq,
                                               const float* __restrict__ bk,
                                               bf16_t* __restrict__ qT,
                                               bf16_t* __restrict__ kT) {
  int wave = (blockIdx.x * blockDim.x + threadIdx.x) >> 5;
  int lane = threadIdx.x & 31;
  int b  = wave / (NPIX / 16);
  int n0 = (wave % (NPIX / 16)) * 16;
  const bf16_t* aBase = xT + ((size_t)b * NPIX + n0) * CH;

  v8f aq0 = zero8(), aq1 = zero8(), ak0 = zero8(), ak1 = zero8();
#pragma unroll
  for (int kc = 0; kc < CH; kc += 32) {
    v16bf a   = load_a_rm(aBase + kc, CH, lane);
    v16bf bq0 = load_b_cm(wq + kc, CH, lane);
    v16bf bq1 = load_b_cm(wq + (size_t)16 * CH + kc, CH, lane);
    v16bf bk0 = load_b_cm(wk + kc, CH, lane);
    v16bf bk1 = load_b_cm(wk + (size_t)16 * CH + kc, CH, lane);
    aq0 = WMMA_BF16(a, bq0, aq0);
    aq1 = WMMA_BF16(a, bq1, aq1);
    ak0 = WMMA_BF16(a, bk0, ak0);
    ak1 = WMMA_BF16(a, bk1, ak1);
  }
  int lr = lane & 15, hi = lane >> 4;
  float biasq0 = bq[lr], biasq1 = bq[16 + lr];
  float biask0 = bk[lr], biask1 = bk[16 + lr];
  bf16_t* qb = qT + (size_t)b * NPIX * CQ;
  bf16_t* kb = kT + (size_t)b * NPIX * CQ;
#pragma unroll
  for (int r = 0; r < 8; ++r) {
    int n = n0 + r + hi * 8;
    qb[(size_t)n * CQ + lr]      = f2bf(fmaxf(aq0[r] + biasq0, 0.0f));
    qb[(size_t)n * CQ + 16 + lr] = f2bf(fmaxf(aq1[r] + biasq1, 0.0f));
    kb[(size_t)n * CQ + lr]      = f2bf(fmaxf(ak0[r] + biask0, 0.0f));
    kb[(size_t)n * CQ + 16 + lr] = f2bf(fmaxf(ak1[r] + biask1, 0.0f));
  }
}

// ---------------------------------------------------------------- v proj
__global__ void __launch_bounds__(256) proj_v(const bf16_t* __restrict__ xT,
                                              const bf16_t* __restrict__ wv,
                                              const float* __restrict__ bv,
                                              bf16_t* __restrict__ vb) {
  int wave = (blockIdx.x * blockDim.x + threadIdx.x) >> 5;
  int lane = threadIdx.x & 31;
  int nb = wave % (NPIX / 64);
  int t  = wave / (NPIX / 64);
  int ct = t % (CH / 16);
  int b  = t / (CH / 16);
  int c0 = ct * 16, n0 = nb * 64;

  const bf16_t* xb = xT + (size_t)b * NPIX * CH;
  v8f acc[4] = {zero8(), zero8(), zero8(), zero8()};
#pragma unroll
  for (int kc = 0; kc < CH; kc += 32) {
    v16bf a = load_a_rm(wv + (size_t)c0 * CH + kc, CH, lane);
#pragma unroll
    for (int j = 0; j < 4; ++j) {
      v16bf bf = load_b_cm(xb + (size_t)(n0 + j * 16) * CH + kc, CH, lane);
      acc[j] = WMMA_BF16(a, bf, acc[j]);
    }
  }
  int lr = lane & 15, hi = lane >> 4;
  bf16_t* vbB = vb + (size_t)b * CH * NPIX;
#pragma unroll
  for (int r = 0; r < 8; ++r) {
    int c = c0 + r + hi * 8;
    float bias = bv[c];
#pragma unroll
    for (int j = 0; j < 4; ++j)
      vbB[(size_t)c * NPIX + n0 + j * 16 + lr] = f2bf(fmaxf(acc[j][r] + bias, 0.0f));
  }
}

// ---------------------------------------------------------------- energy
__global__ void __launch_bounds__(256) energy_gemm(const bf16_t* __restrict__ qT,
                                                   const bf16_t* __restrict__ kT,
                                                   float* __restrict__ att) {
  int wave = (blockIdx.x * blockDim.x + threadIdx.x) >> 5;
  int lane = threadIdx.x & 31;
  int mb = wave % 16;
  int t  = wave / 16;
  int nt = t % (NPIX / 16);
  int b  = t / (NPIX / 16);
  int n0 = nt * 16, m0 = mb * 256;

  v16bf a = load_a_rm(qT + ((size_t)b * NPIX + n0) * CQ, CQ, lane);
  const bf16_t* kB = kT + (size_t)b * NPIX * CQ;
  float* attB = att + (size_t)b * NPIX * NPIX;
  int lr = lane & 15, hi = lane >> 4;
#pragma unroll 4
  for (int j = 0; j < 16; ++j) {
    v16bf bf = load_b_cm(kB + (size_t)(m0 + j * 16) * CQ, CQ, lane);
    v8f acc = zero8();
    acc = WMMA_BF16(a, bf, acc);
#pragma unroll
    for (int r = 0; r < 8; ++r)
      attB[(size_t)(n0 + r + hi * 8) * NPIX + m0 + j * 16 + lr] = acc[r];
  }
}

// ---------------------------------------------------------------- softmax
__global__ void __launch_bounds__(256) softmax_rows(float* __restrict__ att) {
  int wave = (blockIdx.x * blockDim.x + threadIdx.x) >> 5;
  int lane = threadIdx.x & 31;
  int b = wave / NPIX, n = wave % NPIX;
  v4f* row = reinterpret_cast<v4f*>(att + ((size_t)b * NPIX + n) * NPIX);

  float mx = -3.402823466e38f;
  for (int i = lane; i < NPIX / 4; i += 32) {
    v4f t = row[i];
    mx = fmaxf(mx, fmaxf(fmaxf(t[0], t[1]), fmaxf(t[2], t[3])));
  }
#pragma unroll
  for (int o = 16; o > 0; o >>= 1) mx = fmaxf(mx, __shfl_xor(mx, o, 32));

  float s = 0.0f;
  for (int i = lane; i < NPIX / 4; i += 32) {
    v4f t = row[i];
    s += __expf(t[0] - mx) + __expf(t[1] - mx) + __expf(t[2] - mx) + __expf(t[3] - mx);
  }
#pragma unroll
  for (int o = 16; o > 0; o >>= 1) s += __shfl_xor(s, o, 32);
  float inv = 1.0f / s;

  for (int i = lane; i < NPIX / 4; i += 32) {
    v4f t = row[i], w;
    w[0] = __expf(t[0] - mx) * inv; w[1] = __expf(t[1] - mx) * inv;
    w[2] = __expf(t[2] - mx) * inv; w[3] = __expf(t[3] - mx) * inv;
    row[i] = w;
  }
}

// ---------------------------------------------------------------- out GEMM
// out[b][c][n] = gamma * sum_m v[c][m]*att[n][m] + x[b][c][n]
// WG: 256 c-rows x 32 n-cols. The 32x32 fp32 attention tile is DMA'd by the
// Tensor Data Mover into LDS (double buffered, TENSORcnt-tracked, padded to a
// 36-float row stride for conflict-free fragment reads). Wave 0 drives the
// TDM; barriers publish buffers to all 8 waves.
__global__ void __launch_bounds__(256) out_gemm(const bf16_t* __restrict__ vb,
                                                const float* __restrict__ att,
                                                const float* __restrict__ x,
                                                const float* __restrict__ gamma,
                                                float* __restrict__ out) {
#define LDSTRIDE 36
  __shared__ __attribute__((aligned(16))) float sAtt[2][32 * LDSTRIDE];
  int b  = blockIdx.x / (NPIX / 32);
  int n0 = (blockIdx.x % (NPIX / 32)) * 32;
  int tid = threadIdx.x, lane = tid & 31, wave = tid >> 5;
  int c0 = wave * 32;

  const bf16_t* vB  = vb + (size_t)b * CH * NPIX;
  const float* attB = att + (size_t)b * NPIX * NPIX;

  v8f acc[4] = {zero8(), zero8(), zero8(), zero8()};

#if HAVE_TDM
  if (wave == 0)
    tdm_load_tile(attB + (size_t)n0 * NPIX, (unsigned)(size_t)&sAtt[0][0]);
#endif

  for (int kb = 0; kb < NPIX; kb += 32) {
    int cur = (kb >> 5) & 1;
#if HAVE_TDM
    if (wave == 0) __builtin_amdgcn_s_wait_tensorcnt(0);   // cur buffer filled
    __syncthreads();                                       // publish cur
    if (wave == 0 && kb + 32 < NPIX)                       // overlap next DMA
      tdm_load_tile(attB + (size_t)n0 * NPIX + kb + 32,
                    (unsigned)(size_t)&sAtt[cur ^ 1][0]);
#else
    // Fallback: cooperative staged copy (no TDM builtin on this toolchain).
#pragma unroll
    for (int i = 0; i < 4; ++i) {
      int idx = tid + i * 256;
      int rr = idx >> 5, cc = idx & 31;
      sAtt[cur][rr * LDSTRIDE + cc] = attB[(size_t)(n0 + rr) * NPIX + kb + cc];
    }
    __syncthreads();
#endif

    v16bf a0 = load_a_rm(vB + (size_t)c0 * NPIX + kb, NPIX, lane);
    v16bf a1 = load_a_rm(vB + (size_t)(c0 + 16) * NPIX + kb, NPIX, lane);
    v16bf b0 = load_b_cm_f32(sAtt[cur], LDSTRIDE, lane);
    v16bf b1 = load_b_cm_f32(sAtt[cur] + 16 * LDSTRIDE, LDSTRIDE, lane);
    acc[0] = WMMA_BF16(a0, b0, acc[0]);
    acc[1] = WMMA_BF16(a0, b1, acc[1]);
    acc[2] = WMMA_BF16(a1, b0, acc[2]);
    acc[3] = WMMA_BF16(a1, b1, acc[3]);
    __syncthreads();   // all reads of cur done before it is overwritten
  }

  float g = gamma[0];
  int lr = lane & 15, hi = lane >> 4;
#pragma unroll
  for (int mt = 0; mt < 2; ++mt)
#pragma unroll
    for (int nt = 0; nt < 2; ++nt) {
      v8f a = acc[mt * 2 + nt];
#pragma unroll
      for (int r = 0; r < 8; ++r) {
        int c = c0 + mt * 16 + r + hi * 8;
        int n = n0 + nt * 16 + lr;
        size_t idx = ((size_t)b * CH + c) * NPIX + n;
        out[idx] = g * a[r] + x[idx];
      }
    }
#undef LDSTRIDE
}

// ---------------------------------------------------------------- launch
extern "C" void kernel_launch(void* const* d_in, const int* in_sizes, int n_in,
                              void* d_out, int out_size, void* d_ws, size_t ws_size,
                              hipStream_t stream) {
  const float* x     = (const float*)d_in[0];
  const float* Wq    = (const float*)d_in[1];
  const float* bq    = (const float*)d_in[2];
  const float* Wk    = (const float*)d_in[3];
  const float* bk    = (const float*)d_in[4];
  const float* Wv    = (const float*)d_in[5];
  const float* bv    = (const float*)d_in[6];
  const float* gamma = (const float*)d_in[7];

  float* out = (float*)d_out;
  float* att = out + (size_t)BATCH * CH * NPIX;   // attention output region

  bf16_t* xT  = (bf16_t*)d_ws;                    // (B,N,C)
  bf16_t* wqb = xT + (size_t)BATCH * NPIX * CH;
  bf16_t* wkb = wqb + (size_t)CQ * CH;
  bf16_t* wvb = wkb + (size_t)CQ * CH;
  bf16_t* qT  = wvb + (size_t)CH * CH;            // (B,N,CQ)
  bf16_t* kT  = qT + (size_t)BATCH * NPIX * CQ;   // (B,N,CQ)
  bf16_t* vbf = kT + (size_t)BATCH * NPIX * CQ;   // (B,C,N)

  cvt_f32_bf16<<<(CQ * CH + 255) / 256, 256, 0, stream>>>(Wq, wqb, CQ * CH);
  cvt_f32_bf16<<<(CQ * CH + 255) / 256, 256, 0, stream>>>(Wk, wkb, CQ * CH);
  cvt_f32_bf16<<<(CH * CH + 255) / 256, 256, 0, stream>>>(Wv, wvb, CH * CH);

  dim3 tb(32, 8);
  dim3 tg(NPIX / 32, CH / 32, BATCH);
  transpose_cvt<<<tg, tb, 0, stream>>>(x, xT);

  proj_qk<<<BATCH * (NPIX / 16) / 8, 256, 0, stream>>>(xT, wqb, wkb, bq, bk, qT, kT);
  proj_v<<<BATCH * (CH / 16) * (NPIX / 64) / 8, 256, 0, stream>>>(xT, wvb, bv, vbf);
  energy_gemm<<<BATCH * (NPIX / 16) * 16 / 8, 256, 0, stream>>>(qT, kT, att);
  softmax_rows<<<BATCH * NPIX / 8, 256, 0, stream>>>(att);
  out_gemm<<<BATCH * (NPIX / 32), 256, 0, stream>>>(vbf, att, x, gamma, out);
}
